// InterImageTokenizer_44117904064920
// MI455X (gfx1250) — matrix operand order repaired and checked by reading, not compiled, not run on hardware
//
#include <hip/hip_runtime.h>

// ---------------------------------------------------------------------------
// InterImageTokenizer for MI455X (gfx1250, wave32, WMMA)
//   B=64, C=3, HW=384, p=16 -> N=576 patches, D=C*p*p=768, Hd=768, V=1000
//   M = B*N = 36864 rows, K = 768, GEMM1 N=768, GEMM2 N=1024 (vocab padded)
// ---------------------------------------------------------------------------

typedef __attribute__((ext_vector_type(16))) __bf16 v16bf;
typedef __attribute__((ext_vector_type(8)))  float  v8f;

union FragBF { uint4 u[2]; v16bf v; };

#define BATCH   64
#define NPATCH  576
#define HD      768
#define DK      768
#define VOC     1000
#define VOCP    1024
#define MROWS   36864   // BATCH*NPATCH
#define TOK     577     // 1+NPATCH

static __device__ __forceinline__ unsigned short f2bf(float f) {
  unsigned int u = __float_as_uint(f);
  u += 0x7FFFu + ((u >> 16) & 1u);        // round-to-nearest-even
  return (unsigned short)(u >> 16);
}

// ---------------------------------------------------------------------------
// K1: patch extraction -> bf16 patches [M][768] + per-row 1/||patch|| (f32)
// ---------------------------------------------------------------------------
__global__ __launch_bounds__(256) void k_patchify(const float* __restrict__ px,
                                                  unsigned short* __restrict__ patches,
                                                  float* __restrict__ invn) {
  const int row = blockIdx.x;                 // [0, 36864)
  const int b = row / NPATCH, pi = row % NPATCH;
  const int hp = pi / 24, wp = pi % 24;
  const int t = threadIdx.x;                  // 256 threads
  const int ph = t >> 4, pw = t & 15;
  float ss = 0.f;
  #pragma unroll
  for (int c = 0; c < 3; ++c) {
    size_t idx = (((size_t)(b * 3 + c) * 384 + hp * 16 + ph) * 384) + wp * 16 + pw;
    float x = px[idx];
    ss += x * x;
    patches[(size_t)row * DK + c * 256 + t] = f2bf(x);
  }
  __shared__ float red[256];
  red[t] = ss; __syncthreads();
  for (int s = 128; s > 0; s >>= 1) { if (t < s) red[t] += red[t + s]; __syncthreads(); }
  if (t == 0) invn[row] = rsqrtf(red[0]);
}

// ---------------------------------------------------------------------------
// K2: normalize vocab rows -> bf16 vn [1024][768] (rows >= 1000 zeroed)
// ---------------------------------------------------------------------------
__global__ __launch_bounds__(256) void k_normvocab(const float* __restrict__ vocab,
                                                   unsigned short* __restrict__ vn) {
  const int row = blockIdx.x;                 // [0, 1024)
  const int t = threadIdx.x;
  const bool real = row < VOC;
  float x[3] = {0.f, 0.f, 0.f};
  float ss = 0.f;
  if (real) {
    #pragma unroll
    for (int k = 0; k < 3; ++k) { x[k] = vocab[(size_t)row * DK + t + 256 * k]; ss += x[k] * x[k]; }
  }
  __shared__ float red[256];
  red[t] = ss; __syncthreads();
  for (int s = 128; s > 0; s >>= 1) { if (t < s) red[t] += red[t + s]; __syncthreads(); }
  __shared__ float sInv;
  if (t == 0) sInv = real ? rsqrtf(red[0]) : 0.f;
  __syncthreads();
  const float inv = sInv;
  #pragma unroll
  for (int k = 0; k < 3; ++k) vn[(size_t)row * DK + t + 256 * k] = f2bf(x[k] * inv);
}

// ---------------------------------------------------------------------------
// K3: conv_w f32 -> bf16 [768][768]
// ---------------------------------------------------------------------------
__global__ __launch_bounds__(256) void k_cvtW(const float* __restrict__ wsrc,
                                              unsigned short* __restrict__ wb) {
  const int row = blockIdx.x, t = threadIdx.x;
  #pragma unroll
  for (int k = 0; k < 3; ++k) {
    size_t i = (size_t)row * DK + t + 256 * k;
    wb[i] = f2bf(wsrc[i]);
  }
}

// ---------------------------------------------------------------------------
// GEMM1: emb[M][768] = patches @ W^T + bias + pos_embed   (bf16 WMMA, f32 acc)
// block = 256 thr = 8 waves, tile 128x128, BK=32; wave w owns rows [16w,16w+16)
// ---------------------------------------------------------------------------
__global__ __launch_bounds__(256) void k_gemm_emb(const unsigned short* __restrict__ A,
                                                  const unsigned short* __restrict__ Bw,
                                                  const float* __restrict__ bias,
                                                  const float* __restrict__ pos,
                                                  float* __restrict__ emb) {
  __shared__ unsigned short sA[128][40];
  __shared__ unsigned short sB[128][40];
  const int tid = threadIdx.x;
  const int wv = tid >> 5, lane = tid & 31, hi = lane >> 4, ln = lane & 15;
  const int m0 = blockIdx.x * 128, n0 = blockIdx.y * 128;

  const v8f vzero = {0.f, 0.f, 0.f, 0.f, 0.f, 0.f, 0.f, 0.f};
  v8f acc[8];
  #pragma unroll
  for (int a = 0; a < 8; ++a) acc[a] = vzero;

  const int sr = tid >> 1, soff = (tid & 1) * 16;
  for (int kt = 0; kt < 24; ++kt) {
    const int k0 = kt * 32;
    __syncthreads();
    { const uint4* pa = (const uint4*)(A + (size_t)(m0 + sr) * DK + k0 + soff);
      *(uint4*)&sA[sr][soff] = pa[0]; *(uint4*)&sA[sr][soff + 8] = pa[1];
      const uint4* pb = (const uint4*)(Bw + (size_t)(n0 + sr) * DK + k0 + soff);
      *(uint4*)&sB[sr][soff] = pb[0]; *(uint4*)&sB[sr][soff + 8] = pb[1]; }
    __syncthreads();
    FragBF fa;   // A frag: row = 16w+ln, 16 contiguous K at 16*hi
    fa.u[0] = *(const uint4*)&sA[16 * wv + ln][hi * 16];
    fa.u[1] = *(const uint4*)&sA[16 * wv + ln][hi * 16 + 8];
    #pragma unroll
    for (int a = 0; a < 8; ++a) {
      FragBF fb; // B frag: col = 16a+ln, 16 contiguous K at 16*hi
      const int c = a * 16 + ln;
      fb.u[0] = *(const uint4*)&sB[c][hi * 16];
      fb.u[1] = *(const uint4*)&sB[c][hi * 16 + 8];
      acc[a] = __builtin_amdgcn_wmma_f32_16x16x32_bf16(false, fa.v, false, fb.v,
                                                       (short)0, acc[a], false, false);
    }
  }
  // epilogue: + bias + pos_embed[1 + row%576]
  #pragma unroll
  for (int a = 0; a < 8; ++a) {
    const int col = n0 + a * 16 + ln;
    const float bcol = bias[col];
    #pragma unroll
    for (int r = 0; r < 8; ++r) {
      const int row = m0 + 16 * wv + r + 8 * hi;
      const int tok = row % NPATCH;
      emb[(size_t)row * HD + col] = acc[a][r] + bcol + pos[(size_t)(1 + tok) * HD + col];
    }
  }
}

// ---------------------------------------------------------------------------
// GEMM2: q = patches @ vn^T, fused per-row (max, argmax) over 1024 cols
// (argmin of cosine distance == argmax of raw dot with normalized vocab)
// ---------------------------------------------------------------------------
__global__ __launch_bounds__(256) void k_gemm_argmax(const unsigned short* __restrict__ A,
                                                     const unsigned short* __restrict__ Vn,
                                                     float* __restrict__ qmax,
                                                     int* __restrict__ qarg) {
  __shared__ unsigned short sA[128][40];
  __shared__ unsigned short sB[128][40];
  const int tid = threadIdx.x;
  const int wv = tid >> 5, lane = tid & 31, hi = lane >> 4, ln = lane & 15;
  const int m0 = blockIdx.x * 128;
  const int sr = tid >> 1, soff = (tid & 1) * 16;
  const v8f vzero = {0.f, 0.f, 0.f, 0.f, 0.f, 0.f, 0.f, 0.f};
  const float NEGINF = -3.402823466e38f;

  float rowbest[8]; int rowarg[8];
  #pragma unroll
  for (int r = 0; r < 8; ++r) { rowbest[r] = NEGINF; rowarg[r] = 0; }

  for (int nb = 0; nb < 8; ++nb) {
    const int n0 = nb * 128;
    v8f acc[8];
    #pragma unroll
    for (int a = 0; a < 8; ++a) acc[a] = vzero;

    for (int kt = 0; kt < 24; ++kt) {
      const int k0 = kt * 32;
      __syncthreads();
      { const uint4* pa = (const uint4*)(A + (size_t)(m0 + sr) * DK + k0 + soff);
        *(uint4*)&sA[sr][soff] = pa[0]; *(uint4*)&sA[sr][soff + 8] = pa[1];
        const uint4* pb = (const uint4*)(Vn + (size_t)(n0 + sr) * DK + k0 + soff);
        *(uint4*)&sB[sr][soff] = pb[0]; *(uint4*)&sB[sr][soff + 8] = pb[1]; }
      __syncthreads();
      FragBF fa;
      fa.u[0] = *(const uint4*)&sA[16 * wv + ln][hi * 16];
      fa.u[1] = *(const uint4*)&sA[16 * wv + ln][hi * 16 + 8];
      #pragma unroll
      for (int a = 0; a < 8; ++a) {
        FragBF fb;
        const int c = a * 16 + ln;
        fb.u[0] = *(const uint4*)&sB[c][hi * 16];
        fb.u[1] = *(const uint4*)&sB[c][hi * 16 + 8];
        acc[a] = __builtin_amdgcn_wmma_f32_16x16x32_bf16(false, fa.v, false, fb.v,
                                                         (short)0, acc[a], false, false);
      }
    }
    // reduce this 128-column block into running per-row max/argmax
    #pragma unroll
    for (int r = 0; r < 8; ++r) {
      float bv = NEGINF; int bc = 0;
      #pragma unroll
      for (int a = 0; a < 8; ++a) {
        const int col = n0 + a * 16 + ln;
        const float v = (col < VOC) ? acc[a][r] : NEGINF;
        if (v > bv) { bv = v; bc = col; }
      }
      #pragma unroll
      for (int off = 1; off < 16; off <<= 1) {     // reduce within 16-lane half
        float ov = __shfl_xor(bv, off, 32);
        int   oc = __shfl_xor(bc, off, 32);
        if (ov > bv || (ov == bv && oc < bc)) { bv = ov; bc = oc; }
      }
      if (bv > rowbest[r] || (bv == rowbest[r] && bc < rowarg[r])) { rowbest[r] = bv; rowarg[r] = bc; }
    }
  }
  if (ln == 0) {
    #pragma unroll
    for (int r = 0; r < 8; ++r) {
      const int row = m0 + 16 * wv + r + 8 * hi;
      qmax[row] = rowbest[r];
      qarg[row] = rowarg[r];
    }
  }
}

// ---------------------------------------------------------------------------
// K5a: per-image count of out-of-codebook patches (smin > 0.1)
// ---------------------------------------------------------------------------
__global__ void k_maskcnt(const float* __restrict__ qmax, const float* __restrict__ invn,
                          int* __restrict__ cnt) {
  const int b = blockIdx.x, n = threadIdx.x;   // block = 576 threads
  const int row = b * NPATCH + n;
  const float smin = 1.f - qmax[row] * invn[row];
  __shared__ int c;
  if (n == 0) c = 0;
  __syncthreads();
  if (smin > 0.1f) atomicAdd(&c, 1);
  __syncthreads();
  if (n == 0) cnt[b] = c;
}

// K5b: exclusive prefix over 64 image counts (tiny serial scan)
__global__ void k_prefix(const int* __restrict__ cnt, int* __restrict__ pref) {
  if (threadIdx.x == 0) {
    int s = 0;
    for (int b = 0; b < BATCH; ++b) { pref[b] = s; s += cnt[b]; }
  }
}

// ---------------------------------------------------------------------------
// K6: per-image fresh-id assignment + rank compaction + segment counts
//   block = 576 threads, one image per block
// ---------------------------------------------------------------------------
__global__ void k_labels(const float* __restrict__ qmax, const int* __restrict__ qarg,
                         const float* __restrict__ invn, const int* __restrict__ pref,
                         int* __restrict__ labout, int* __restrict__ counts) {
  const int b = blockIdx.x, n = threadIdx.x;
  const int row = b * NPATCH + n;
  __shared__ int lab[NPATCH];
  __shared__ int sc[NPATCH];
  __shared__ int fo[NPATCH];
  __shared__ int cnts[NPATCH];

  const float smin = 1.f - qmax[row] * invn[row];
  const int mask = (smin > 0.1f) ? 1 : 0;
  sc[n] = mask;
  __syncthreads();
  for (int off = 1; off < NPATCH; off <<= 1) {   // Hillis-Steele inclusive scan
    int v = (n >= off) ? sc[n - off] : 0;
    __syncthreads();
    sc[n] += v;
    __syncthreads();
  }
  const int l = mask ? (VOC + pref[b] + sc[n] - 1) : qarg[row];
  lab[n] = l;
  __syncthreads();
  int f = 1;
  for (int j = 0; j < n; ++j) if (lab[j] == l) { f = 0; break; }
  fo[n] = f;
  __syncthreads();
  int rk = 0;
  for (int j = 0; j < NPATCH; ++j) rk += (fo[j] && (lab[j] < l)) ? 1 : 0;
  labout[row] = rk;
  cnts[n] = 0;
  __syncthreads();
  atomicAdd(&cnts[rk], 1);
  __syncthreads();
  counts[row] = cnts[n];
}

// ---------------------------------------------------------------------------
// K7: build output tokens. grid (577, 64); slot 0 = cls+pos, others = seg mean
//     (deterministic: sequential accumulation over n; no atomics)
// ---------------------------------------------------------------------------
__global__ __launch_bounds__(256) void k_out(const float* __restrict__ emb,
                                             const int* __restrict__ labout,
                                             const int* __restrict__ counts,
                                             const float* __restrict__ cls,
                                             const float* __restrict__ pos,
                                             const float* __restrict__ pad,
                                             float* __restrict__ outp) {
  const int b = blockIdx.y, t = threadIdx.x;
  __shared__ int lab[NPATCH];
  for (int i = t; i < NPATCH; i += 256) lab[i] = labout[b * NPATCH + i];
  __syncthreads();
  const size_t obase = ((size_t)b * TOK + blockIdx.x) * HD;
  if (blockIdx.x == 0) {
    #pragma unroll
    for (int k = 0; k < 3; ++k) { int col = t + 256 * k; outp[obase + col] = cls[col] + pos[col]; }
    return;
  }
  const int s = blockIdx.x - 1;
  const int c = counts[b * NPATCH + s];
  if (c == 0) {
    #pragma unroll
    for (int k = 0; k < 3; ++k) { int col = t + 256 * k; outp[obase + col] = pad[col]; }
    return;
  }
  float a0 = 0.f, a1 = 0.f, a2 = 0.f;
  for (int n = 0; n < NPATCH; ++n) {
    if (lab[n] == s) {
      const float* e = emb + ((size_t)b * NPATCH + n) * HD;
      a0 += e[t]; a1 += e[t + 256]; a2 += e[t + 512];
    }
  }
  const float inv = 1.f / (float)c;
  outp[obase + t] = a0 * inv;
  outp[obase + t + 256] = a1 * inv;
  outp[obase + t + 512] = a2 * inv;
}

// ---------------------------------------------------------------------------
// K8: expanded attention mask [B,1,577,577]; value depends only on key j
// ---------------------------------------------------------------------------
__global__ __launch_bounds__(256) void k_attn(const int* __restrict__ counts,
                                              float* __restrict__ attn) {
  const int i = blockIdx.x, b = blockIdx.y;
  float* base = attn + ((size_t)b * TOK + i) * TOK;
  for (int j = threadIdx.x; j < TOK; j += 256) {
    const bool empty = (j > 0) && (counts[b * NPATCH + (j - 1)] == 0);
    base[j] = empty ? -3.402823466e38f : 0.f;
  }
}

// ---------------------------------------------------------------------------
extern "C" void kernel_launch(void* const* d_in, const int* in_sizes, int n_in,
                              void* d_out, int out_size, void* d_ws, size_t ws_size,
                              hipStream_t stream) {
  (void)in_sizes; (void)n_in; (void)out_size; (void)ws_size;
  const float* px    = (const float*)d_in[0];  // [64,3,384,384]
  const float* convw = (const float*)d_in[1];  // [768,3,16,16] -> [768,768]
  const float* convb = (const float*)d_in[2];  // [768]
  const float* cls   = (const float*)d_in[3];  // [1,1,768]
  const float* pos   = (const float*)d_in[4];  // [1,577,768]
  const float* vocab = (const float*)d_in[5];  // [1000,768]
  const float* pad   = (const float*)d_in[6];  // [1,1,768]

  // d_out regions: out f32 [64,577,768] | labels int32 [64,576] | attn f32 [64,1,577,577]
  float* outp = (float*)d_out;
  int*   labout = (int*)(outp + (size_t)BATCH * TOK * HD);
  float* attnp = (float*)(labout + (size_t)BATCH * NPATCH);

  // workspace carve-up (all offsets multiples of 256 B)
  char* w = (char*)d_ws;
  unsigned short* patches = (unsigned short*)(w);                 // M*768 bf16
  unsigned short* vn      = (unsigned short*)(w + 56623104);      // 1024*768 bf16
  unsigned short* wb      = (unsigned short*)(w + 58195968);      // 768*768 bf16
  float* emb  = (float*)(w + 59375616);                           // M*768 f32
  float* invn = (float*)(w + 172621824);                          // M f32
  float* qmax = (float*)(w + 172769280);                          // M f32
  int*   qarg = (int*)  (w + 172916736);                          // M i32
  int*   cnt  = (int*)  (w + 173064192);                          // 64 i32
  int*   pref = (int*)  (w + 173064448);                          // 64 i32
  int*   counts = (int*)(w + 173064704);                          // M i32 (per-image seg counts)

  k_patchify<<<MROWS, 256, 0, stream>>>(px, patches, invn);
  k_normvocab<<<VOCP, 256, 0, stream>>>(vocab, vn);
  k_cvtW<<<HD, 256, 0, stream>>>(convw, wb);

  k_gemm_emb<<<dim3(MROWS / 128, HD / 128), 256, 0, stream>>>(patches, wb, convb, pos, emb);
  k_gemm_argmax<<<MROWS / 128, 256, 0, stream>>>(patches, vn, qmax, qarg);

  k_maskcnt<<<BATCH, NPATCH, 0, stream>>>(qmax, invn, cnt);
  k_prefix<<<1, 64, 0, stream>>>(cnt, pref);
  k_labels<<<BATCH, NPATCH, 0, stream>>>(qmax, qarg, invn, pref, labout, counts);

  k_out<<<dim3(TOK, BATCH), 256, 0, stream>>>(emb, labout, counts, cls, pos, pad, outp);
  k_attn<<<dim3(TOK, BATCH), 256, 0, stream>>>(counts, attnp);
}
